// RoomDescriptor_52424370815601
// MI455X (gfx1250) — matrix operand order repaired and checked by reading, not compiled, not run on hardware
//
#include <hip/hip_runtime.h>
#include <hip/hip_bf16.h>

#define NN    512   // nodes
#define NF    256   // input features
#define HID   64    // per-head hidden
#define NH    8     // heads
#define NO    128   // output features
#define NB    32    // batch

typedef __attribute__((ext_vector_type(16))) __bf16 v16bf;
typedef __attribute__((ext_vector_type(8)))  float  v8f;

// ---------------- helpers ----------------

static __device__ __forceinline__ __bf16 f2bf(float f) {
    return (__bf16)f;   // native convert
}
static __device__ __forceinline__ unsigned short f2bf_bits(float f) {
    __bf16 h = (__bf16)f;
    return __builtin_bit_cast(unsigned short, h);
}
static __device__ __forceinline__ void put2(v16bf& a, int e, unsigned int w) {
    a[2 * e]     = __builtin_bit_cast(__bf16, (unsigned short)(w & 0xffffu));
    a[2 * e + 1] = __builtin_bit_cast(__bf16, (unsigned short)(w >> 16));
}
static __device__ __forceinline__ float leaky(float x, float s) {
    return x >= 0.0f ? x : s * x;
}
// A-matrix (16x32 bf16) K offset for vgpr-pair e (0..7), half-wave g (0/1)
static __device__ __forceinline__ int koffA(int g, int e) {
    return (e < 4) ? (g * 8 + 2 * e) : (16 + g * 8 + 2 * (e - 4));
}
// B-matrix (32x16 bf16) K offset
static __device__ __forceinline__ int koffB(int g, int e) {
    return (e < 4) ? (g * 8 + 2 * e) : (16 + g * 8 + 2 * (e - 4));
}
// reduce across the 16 lanes of a half-wave
static __device__ __forceinline__ float red16(float v) {
    v += __shfl_xor(v, 1, 32);
    v += __shfl_xor(v, 2, 32);
    v += __shfl_xor(v, 4, 32);
    v += __shfl_xor(v, 8, 32);
    return v;
}

#define WMMA_BF16(A, B, C) \
    __builtin_amdgcn_wmma_f32_16x16x32_bf16(false, (A), false, (B), (short)0, (C), false, false)

// ---------------- stage 0a: zero the mean accumulator ----------------
__global__ void zero_kernel(float* __restrict__ p, int n) {
    int i = blockIdx.x * blockDim.x + threadIdx.x;
    if (i < n) p[i] = 0.0f;
}

// ---------------- stage 0b: bulk f32 -> bf16 (pair-packed) ----------------
__global__ void cvt_bf16_kernel(const float* __restrict__ src,
                                unsigned int* __restrict__ dst, int npairs) {
    int i = blockIdx.x * blockDim.x + threadIdx.x;
    if (i < npairs) {
        float2 v = ((const float2*)src)[i];
        dst[i] = (unsigned int)f2bf_bits(v.x) | ((unsigned int)f2bf_bits(v.y) << 16);
    }
}

// ---------------- stage 1: H = X @ W_tran^T, e1/e2 ----------------
// grid = NB*NH*(NN/16), block = 32 (one wave per 16-row strip of one head)
__global__ __launch_bounds__(32) void gat_h_kernel(
    const unsigned int* __restrict__ Xbf,   // [NB, NN, NF] bf16 pairs
    const unsigned int* __restrict__ Wtbf,  // [NH, HID, NF] bf16 pairs
    const float* __restrict__ a1,           // [NH, HID]
    const float* __restrict__ a2,           // [NH, HID]
    unsigned short* __restrict__ Ht,        // [NB*NH, HID, NN] bf16 (transposed)
    float* __restrict__ e1,                 // [NB*NH, NN]
    float* __restrict__ e2)                 // [NB*NH, NN]
{
    const int lane = threadIdx.x;
    const int jt = blockIdx.x & 31;
    const int bh = blockIdx.x >> 5;
    const int h  = bh & (NH - 1);
    const int b  = bh >> 3;
    const int g = lane >> 4, n = lane & 15;
    const int jbase = jt * 16;

    const unsigned int* ap = Xbf + (((size_t)b * NN + jbase + n) * NF >> 1);
    v8f c[4] = {};

    for (int ks = 0; ks < NF; ks += 32) {
        v16bf a;
#pragma unroll
        for (int e = 0; e < 8; e++) put2(a, e, ap[(ks + koffA(g, e)) >> 1]);
#pragma unroll
        for (int t = 0; t < 4; t++) {
            const unsigned int* wp = Wtbf + (((size_t)h * HID + 16 * t + n) * NF >> 1);
            v16bf bt;
#pragma unroll
            for (int e = 0; e < 8; e++) put2(bt, e, wp[(ks + koffB(g, e)) >> 1]);
            c[t] = WMMA_BF16(a, bt, c[t]);
        }
    }

    // store H transposed as bf16: Ht[bh][d][node]
#pragma unroll
    for (int t = 0; t < 4; t++) {
#pragma unroll
        for (int r = 0; r < 8; r++) {
            int node = jbase + r + 8 * g;
            int d = 16 * t + n;
            Ht[((size_t)bh * HID + d) * NN + node] = f2bf_bits(c[t][r]);
        }
    }

    // e1/e2 from f32 accumulators
    float a1v[4], a2v[4];
#pragma unroll
    for (int t = 0; t < 4; t++) {
        a1v[t] = a1[h * HID + 16 * t + n];
        a2v[t] = a2[h * HID + 16 * t + n];
    }
#pragma unroll
    for (int r = 0; r < 8; r++) {
        float s1 = 0.0f, s2 = 0.0f;
#pragma unroll
        for (int t = 0; t < 4; t++) {
            s1 += c[t][r] * a1v[t];
            s2 += c[t][r] * a2v[t];
        }
        s1 = red16(s1);
        s2 = red16(s2);
        if (n == 0) {
            int node = jbase + 8 * g + r;
            e1[(size_t)bh * NN + node] = s1;
            e2[(size_t)bh * NN + node] = s2;
        }
    }
}

// ---------------- stage 2: attention, concat, leaky(0.01) -> x2 ----------------
// grid = NB*NH*(NN/16), block = 32
__global__ __launch_bounds__(32) void gat_attn_kernel(
    const float* __restrict__ e1,
    const float* __restrict__ e2,
    const unsigned short* __restrict__ Ht,  // [NB*NH, HID, NN] bf16
    unsigned short* __restrict__ x2)        // [NB, NN, NH*HID] bf16
{
    __shared__ float e1s[NN];
    __shared__ float rsum[16];

    const int lane = threadIdx.x;
    const int jt = blockIdx.x & 31;
    const int bh = blockIdx.x >> 5;
    const int h  = bh & (NH - 1);
    const int b  = bh >> 3;
    const int g = lane >> 4, n = lane & 15;
    const int jbase = jt * 16;

    const float* e1p = e1 + (size_t)bh * NN;
    float mx = -3.0e38f;
    for (int i = lane; i < NN; i += 32) {
        float v = e1p[i];
        e1s[i] = v;
        mx = fmaxf(mx, v);
    }
#pragma unroll
    for (int m = 1; m < 32; m <<= 1) mx = fmaxf(mx, __shfl_xor(mx, m, 32));
    __syncthreads();

    const float e2v = e2[(size_t)bh * NN + jbase + n];   // row j = jbase + n
    const float mrow = leaky(mx + e2v, 0.2f);            // exact row max (leaky monotone)

    float rs = 0.0f;
    v8f c[4] = {};

    for (int ks = 0; ks < NN; ks += 32) {
        v16bf a;
#pragma unroll
        for (int e = 0; e < 8; e++) {
            int k = ks + koffA(g, e);
            float p0 = __expf(leaky(e1s[k] + e2v, 0.2f) - mrow);
            float p1 = __expf(leaky(e1s[k + 1] + e2v, 0.2f) - mrow);
            rs += p0 + p1;
            a[2 * e] = f2bf(p0);
            a[2 * e + 1] = f2bf(p1);
        }
#pragma unroll
        for (int t = 0; t < 4; t++) {
            const unsigned int* hp =
                (const unsigned int*)(Ht + ((size_t)bh * HID + 16 * t + n) * NN);
            v16bf bt;
#pragma unroll
            for (int e = 0; e < 8; e++) put2(bt, e, hp[(ks + koffB(g, e)) >> 1]);
            c[t] = WMMA_BF16(a, bt, c[t]);
        }
    }

    rs += __shfl_xor(rs, 16, 32);          // combine complementary K halves of row n
    if (lane < 16) rsum[lane] = rs;
    __syncthreads();

#pragma unroll
    for (int t = 0; t < 4; t++) {
#pragma unroll
        for (int r = 0; r < 8; r++) {
            int node = jbase + r + 8 * g;
            float val = c[t][r] / rsum[r + 8 * g];
            val = leaky(val, 0.01f);       // acvt LeakyReLU after head-concat
            x2[((size_t)b * NN + node) * (NH * HID) + h * HID + 16 * t + n] = f2bf_bits(val);
        }
    }
}

// ---------------- stage 3: h2 = x2 @ W_out^T, f1/f2 ----------------
// grid = NB*(NN/16), block = 32
__global__ __launch_bounds__(32) void out_gemm_kernel(
    const unsigned short* __restrict__ x2,    // [NB, NN, 512] bf16
    const unsigned int* __restrict__ Woutbf,  // [NO, 512] bf16 pairs
    const float* __restrict__ a1o,            // [NO]
    const float* __restrict__ a2o,            // [NO]
    unsigned short* __restrict__ h2t,         // [NB, NO, NN] bf16 (transposed)
    float* __restrict__ f1,                   // [NB, NN]
    float* __restrict__ f2)                   // [NB, NN]
{
    const int lane = threadIdx.x;
    const int jt = blockIdx.x & 31;
    const int b  = blockIdx.x >> 5;
    const int g = lane >> 4, n = lane & 15;
    const int jbase = jt * 16;
    const int K = NH * HID;  // 512

    const unsigned int* ap =
        (const unsigned int*)(x2 + ((size_t)b * NN + jbase + n) * K);

    v8f c[8] = {};
    for (int ks = 0; ks < K; ks += 32) {
        v16bf a;
#pragma unroll
        for (int e = 0; e < 8; e++) put2(a, e, ap[(ks + koffA(g, e)) >> 1]);
#pragma unroll
        for (int t = 0; t < 8; t++) {
            const unsigned int* wp = Woutbf + ((size_t)(16 * t + n) * K >> 1);
            v16bf bt;
#pragma unroll
            for (int e = 0; e < 8; e++) put2(bt, e, wp[(ks + koffB(g, e)) >> 1]);
            c[t] = WMMA_BF16(a, bt, c[t]);
        }
    }

#pragma unroll
    for (int t = 0; t < 8; t++) {
#pragma unroll
        for (int r = 0; r < 8; r++) {
            int node = jbase + r + 8 * g;
            int o = 16 * t + n;
            h2t[((size_t)b * NO + o) * NN + node] = f2bf_bits(c[t][r]);
        }
    }

    float a1v[8], a2v[8];
#pragma unroll
    for (int t = 0; t < 8; t++) {
        a1v[t] = a1o[16 * t + n];
        a2v[t] = a2o[16 * t + n];
    }
#pragma unroll
    for (int r = 0; r < 8; r++) {
        float s1 = 0.0f, s2 = 0.0f;
#pragma unroll
        for (int t = 0; t < 8; t++) {
            s1 += c[t][r] * a1v[t];
            s2 += c[t][r] * a2v[t];
        }
        s1 = red16(s1);
        s2 = red16(s2);
        if (n == 0) {
            int node = jbase + 8 * g + r;
            f1[(size_t)b * NN + node] = s1;
            f2[(size_t)b * NN + node] = s2;
        }
    }
}

// ---------------- stage 4: output attention + mean over rows ----------------
// grid = NB*(NN/16), block = 32; accumulates mean[b][o] via atomics
__global__ __launch_bounds__(32) void final_attn_kernel(
    const float* __restrict__ f1,
    const float* __restrict__ f2,
    const unsigned short* __restrict__ h2t,  // [NB, NO, NN] bf16
    float* __restrict__ meanbuf)             // [NB, NO] (pre-zeroed)
{
    __shared__ float f1s[NN];
    __shared__ float rsum[16];

    const int lane = threadIdx.x;
    const int jt = blockIdx.x & 31;
    const int b  = blockIdx.x >> 5;
    const int g = lane >> 4, n = lane & 15;
    const int jbase = jt * 16;

    const float* f1p = f1 + (size_t)b * NN;
    float mx = -3.0e38f;
    for (int i = lane; i < NN; i += 32) {
        float v = f1p[i];
        f1s[i] = v;
        mx = fmaxf(mx, v);
    }
#pragma unroll
    for (int m = 1; m < 32; m <<= 1) mx = fmaxf(mx, __shfl_xor(mx, m, 32));
    __syncthreads();

    const float f2v = f2[(size_t)b * NN + jbase + n];
    const float mrow = leaky(mx + f2v, 0.2f);

    float rs = 0.0f;
    v8f c[8] = {};

    for (int ks = 0; ks < NN; ks += 32) {
        v16bf a;
#pragma unroll
        for (int e = 0; e < 8; e++) {
            int k = ks + koffA(g, e);
            float p0 = __expf(leaky(f1s[k] + f2v, 0.2f) - mrow);
            float p1 = __expf(leaky(f1s[k + 1] + f2v, 0.2f) - mrow);
            rs += p0 + p1;
            a[2 * e] = f2bf(p0);
            a[2 * e + 1] = f2bf(p1);
        }
#pragma unroll
        for (int t = 0; t < 8; t++) {
            const unsigned int* hp =
                (const unsigned int*)(h2t + ((size_t)b * NO + 16 * t + n) * NN);
            v16bf bt;
#pragma unroll
            for (int e = 0; e < 8; e++) put2(bt, e, hp[(ks + koffB(g, e)) >> 1]);
            c[t] = WMMA_BF16(a, bt, c[t]);
        }
    }

    rs += __shfl_xor(rs, 16, 32);
    if (lane < 16) rsum[lane] = rs;
    __syncthreads();

    float inv[8];
#pragma unroll
    for (int r = 0; r < 8; r++) inv[r] = 1.0f / (rsum[r + 8 * g] * (float)NN);

#pragma unroll
    for (int t = 0; t < 8; t++) {
        float part = 0.0f;
#pragma unroll
        for (int r = 0; r < 8; r++) part += c[t][r] * inv[r];
        part += __shfl_xor(part, 16, 32);   // combine the two row halves (same col)
        if (lane < 16) atomicAdd(&meanbuf[(size_t)b * NO + 16 * t + lane], part);
    }
}

// ---------------- stage 5: linear + L2-normalize ----------------
// grid = NB, block = NO
__global__ __launch_bounds__(NO) void lin_norm_kernel(
    const float* __restrict__ meanbuf,   // [NB, NO]
    const float* __restrict__ Wlin,      // [NO, NO]
    const float* __restrict__ blin,      // [NO]
    float* __restrict__ out)             // [NB, NO]
{
    __shared__ float mv[NO];
    __shared__ float red[NO];
    const int b = blockIdx.x;
    const int o = threadIdx.x;

    mv[o] = meanbuf[(size_t)b * NO + o];
    __syncthreads();

    float acc = blin[o];
    for (int f = 0; f < NO; f++) acc += mv[f] * Wlin[(size_t)o * NO + f];

    red[o] = acc * acc;
    __syncthreads();
    for (int s = NO / 2; s > 0; s >>= 1) {
        if (o < s) red[o] += red[o + s];
        __syncthreads();
    }
    float nrm = fmaxf(sqrtf(red[0]), 1e-12f);
    out[(size_t)b * NO + o] = acc / nrm;
}

// ---------------- host launcher ----------------

// workspace layout (byte offsets, all 256B aligned)
#define OFF_E1   ((size_t)0)          // NB*NH*NN f32        = 512 KB
#define OFF_E2   ((size_t)524288)     // 512 KB
#define OFF_F1   ((size_t)1048576)    // NB*NN f32           = 64 KB
#define OFF_F2   ((size_t)1114112)    // 64 KB
#define OFF_MEAN ((size_t)1179648)    // NB*NO f32           = 16 KB
#define OFF_WTB  ((size_t)1196032)    // NH*HID*NF bf16      = 256 KB
#define OFF_WOB  ((size_t)1458176)    // NO*512 bf16         = 128 KB
#define OFF_XB   ((size_t)1589248)    // NB*NN*NF bf16       = 8 MB
#define OFF_HT   ((size_t)9977856)    // NB*NH*HID*NN bf16   = 16 MB
#define OFF_X2   ((size_t)26755072)   // NB*NN*512 bf16      = 16 MB
#define OFF_H2T  ((size_t)43532288)   // NB*NO*NN bf16       = 4 MB
                                      // total ~45.6 MB

extern "C" void kernel_launch(void* const* d_in, const int* in_sizes, int n_in,
                              void* d_out, int out_size, void* d_ws, size_t ws_size,
                              hipStream_t stream) {
    (void)in_sizes; (void)n_in; (void)out_size; (void)ws_size;

    const float* X    = (const float*)d_in[0];  // [NB, NN, NF]
    const float* Wt   = (const float*)d_in[1];  // [NH, HID, NF]
    const float* a1   = (const float*)d_in[2];  // [NH, HID]
    const float* a2   = (const float*)d_in[3];  // [NH, HID]
    const float* Wout = (const float*)d_in[4];  // [NO, 512]
    const float* a1o  = (const float*)d_in[5];  // [NO]
    const float* a2o  = (const float*)d_in[6];  // [NO]
    const float* Wlin = (const float*)d_in[7];  // [NO, NO]
    const float* blin = (const float*)d_in[8];  // [NO]
    float* out = (float*)d_out;                 // [NB, NO]

    char* ws = (char*)d_ws;
    float* e1 = (float*)(ws + OFF_E1);
    float* e2 = (float*)(ws + OFF_E2);
    float* f1 = (float*)(ws + OFF_F1);
    float* f2 = (float*)(ws + OFF_F2);
    float* meanbuf = (float*)(ws + OFF_MEAN);
    unsigned int* Wtbf   = (unsigned int*)(ws + OFF_WTB);
    unsigned int* Woutbf = (unsigned int*)(ws + OFF_WOB);
    unsigned int* Xbf    = (unsigned int*)(ws + OFF_XB);
    unsigned short* Ht   = (unsigned short*)(ws + OFF_HT);
    unsigned short* x2   = (unsigned short*)(ws + OFF_X2);
    unsigned short* h2t  = (unsigned short*)(ws + OFF_H2T);

    // stage 0: zero mean accumulator + one-shot f32->bf16 of GEMM operands
    zero_kernel<<<(NB * NO + 255) / 256, 256, 0, stream>>>(meanbuf, NB * NO);
    cvt_bf16_kernel<<<(NB * NN * NF / 2 + 255) / 256, 256, 0, stream>>>(X, Xbf, NB * NN * NF / 2);
    cvt_bf16_kernel<<<(NH * HID * NF / 2 + 255) / 256, 256, 0, stream>>>(Wt, Wtbf, NH * HID * NF / 2);
    cvt_bf16_kernel<<<(NO * 512 / 2 + 255) / 256, 256, 0, stream>>>(Wout, Woutbf, NO * 512 / 2);

    // stage 1: per-head projection + e1/e2
    gat_h_kernel<<<NB * NH * (NN / 16), 32, 0, stream>>>(Xbf, Wtbf, a1, a2, Ht, e1, e2);

    // stage 2: per-head attention + concat + leaky(0.01)
    gat_attn_kernel<<<NB * NH * (NN / 16), 32, 0, stream>>>(e1, e2, Ht, x2);

    // stage 3: output projection + f1/f2
    out_gemm_kernel<<<NB * (NN / 16), 32, 0, stream>>>(x2, Woutbf, a1o, a2o, h2t, f1, f2);

    // stage 4: output attention + mean over nodes
    final_attn_kernel<<<NB * (NN / 16), 32, 0, stream>>>(f1, f2, h2t, meanbuf);

    // stage 5: final linear + L2 normalize
    lin_norm_kernel<<<NB, NO, 0, stream>>>(meanbuf, Wlin, blin, out);
}
